// UnidirectionalAttention_3058016714918
// MI455X (gfx1250) — matrix operand order repaired and checked by reading, not compiled, not run on hardware
//
#include <hip/hip_runtime.h>

// ---------------------------------------------------------------------------
// Causal multi-head attention forward for MI455X (gfx1250, wave32, WMMA).
// S=4096, HIDDEN=1024, 16 heads x 64. bf16 WMMA (f32 accum) everywhere,
// flash-attention fused middle so the 4096x4096 score tensor never hits HBM.
// ---------------------------------------------------------------------------

#define SEQ    4096
#define HID    1024
#define NH     16
#define HD     64

typedef __attribute__((ext_vector_type(16))) __bf16 v16bf;
typedef __attribute__((ext_vector_type(8)))  __bf16 v8bf;
typedef __attribute__((ext_vector_type(4)))  __bf16 v4bf;
typedef __attribute__((ext_vector_type(8)))  float  v8f;

static __device__ __forceinline__ v8f wmma_bf16(v16bf a, v16bf b, v8f c) {
  // D = A x B + C, 16x16x32 bf16 -> f32
  return __builtin_amdgcn_wmma_f32_16x16x32_bf16(
      /*neg_a=*/false, a, /*neg_b=*/false, b,
      /*c_mod=*/(short)0, c, /*reuse_a=*/false, /*reuse_b=*/false);
}

// A-matrix (16x32 bf16) fragment: lane L holds row m = L%16; element j maps to
// k = j + 16*(L/16). One contiguous 32B run per lane.
static __device__ __forceinline__ v16bf load_a_frag(const __bf16* row, int half) {
  return *(const v16bf*)(row + 16 * half);
}

// B-matrix (32x16 bf16) fragment: lane L = column L%16; element j maps to
// k = j + 16*(L/16). One contiguous 32B run per lane (callers keep 32B align).
static __device__ __forceinline__ v16bf load_b_frag(const __bf16* col, int half) {
  return *(const v16bf*)(col + 16 * half);
}

// ---------------------------------------------------------------------------
// fp32 -> bf16 conversion (x, qkv_w, out_w), 4 elems/thread
// ---------------------------------------------------------------------------
__global__ __launch_bounds__(256) void cvt_bf16(const float* __restrict__ src,
                                                __bf16* __restrict__ dst, int n) {
  int i = (blockIdx.x * 256 + threadIdx.x) * 4;
  if (i < n) {
    float4 f = *(const float4*)(src + i);
    v4bf o;
    o[0] = (__bf16)f.x; o[1] = (__bf16)f.y; o[2] = (__bf16)f.z; o[3] = (__bf16)f.w;
    *(v4bf*)(dst + i) = o;
  }
}

// ---------------------------------------------------------------------------
// QKV projection: qkv = x @ qkv_w^T + b.  Each wave computes a 32(M)x64(N)
// tile. Epilogue scatters into Q[h][s][d]*0.125 (scale folded), K[h][s][d],
// and V transposed Vt[h][d][s] (so PV B-fragments are contiguous later).
// Grid: (4096/32)*(3072/64)/8 = 768 blocks of 256.
// ---------------------------------------------------------------------------
__global__ __launch_bounds__(256) void qkv_gemm(
    const __bf16* __restrict__ xb, const __bf16* __restrict__ wb,
    const float* __restrict__ qkv_b,
    __bf16* __restrict__ Qb, __bf16* __restrict__ Kb, __bf16* __restrict__ Vtb) {
  const int lane = threadIdx.x & 31, wave = threadIdx.x >> 5;
  const int n16 = lane & 15, half = lane >> 4;
  const int wid = blockIdx.x * 8 + wave;
  const int mt = wid / 48, nt = wid % 48;          // 128 x 48 tiles
  const int mbase = mt * 32, nbase = nt * 64;

  v8f acc[2][4] = {};
  for (int k = 0; k < HID; k += 32) {
    v16bf a0 = load_a_frag(xb + (mbase + n16) * HID + k, half);
    v16bf a1 = load_a_frag(xb + (mbase + 16 + n16) * HID + k, half);
#pragma unroll
    for (int j = 0; j < 4; ++j) {
      v16bf b = load_b_frag(wb + (nbase + j * 16 + n16) * HID + k, half);
      acc[0][j] = wmma_bf16(a0, b, acc[0][j]);
      acc[1][j] = wmma_bf16(a1, b, acc[1][j]);
    }
  }

#pragma unroll
  for (int j = 0; j < 4; ++j) {
    const int gibase = nbase + j * 16;             // wave-uniform
    const int part = gibase >> 10;                 // 0=Q 1=K 2=V
    const int hrem = gibase & 1023;
    const int h = hrem >> 6;
    const int d = (hrem & 63) + n16;
    const float bias = qkv_b[gibase + n16];
#pragma unroll
    for (int mi = 0; mi < 2; ++mi) {
#pragma unroll
      for (int r = 0; r < 8; ++r) {
        const int s = mbase + mi * 16 + r + 8 * half;
        const float v = acc[mi][j][r] + bias;
        if (part == 0)      Qb[((h * SEQ + s) << 6) + d] = (__bf16)(v * 0.125f);
        else if (part == 1) Kb[((h * SEQ + s) << 6) + d] = (__bf16)v;
        else                Vtb[((h << 6) + d) * SEQ + s] = (__bf16)v;
      }
    }
  }
}

// ---------------------------------------------------------------------------
// Fused causal flash attention. One wave = 16 query rows of one head; streams
// keys in 32-wide steps: 4 WMMAs for S = Q K^T, online softmax with shfl_xor
// row reductions, LDS transpose of P (C-layout f32 -> A-layout bf16),
// 4 WMMAs for ctx += P V. Writes ctx bf16 [s][h*64+d].
// Grid: 16 heads * 256 qtiles / 8 waves = 512 blocks of 256.
// ---------------------------------------------------------------------------
__global__ __launch_bounds__(256) void flash_attn(
    const __bf16* __restrict__ Qb, const __bf16* __restrict__ Kb,
    const __bf16* __restrict__ Vtb, __bf16* __restrict__ ctxb) {
  __shared__ __attribute__((aligned(32))) __bf16 lds[8][16][32];  // 8KB: P staging
  const int lane = threadIdx.x & 31, wave = threadIdx.x >> 5;
  const int n16 = lane & 15, half = lane >> 4;
  const int flat = blockIdx.x * 8 + wave;
  const int h = flat >> 8;
  const int qb = (flat & 255) << 4;

  const __bf16* qrow = Qb + ((h * SEQ + qb + n16) << 6);
  const v16bf qf0 = load_a_frag(qrow, half);        // d 0..31  (scale pre-folded)
  const v16bf qf1 = load_a_frag(qrow + 32, half);   // d 32..63

  v8f acc[4] = {};
  float mrow[8], lrow[8];
#pragma unroll
  for (int r = 0; r < 8; ++r) { mrow[r] = -1e30f; lrow[r] = 0.f; }

  const int kend = qb + 16;                         // causal: keys <= qb+15
  for (int kb = 0; kb < kend; kb += 32) {
    // ---- S = Q K^T (two 16-key subtiles) ----
    const __bf16* k0 = Kb + ((h * SEQ + kb + n16) << 6);
    const __bf16* k1 = Kb + ((h * SEQ + kb + 16 + n16) << 6);
    v8f z = {};
    v8f s0 = wmma_bf16(qf0, load_b_frag(k0, half), z);
    s0 = wmma_bf16(qf1, load_b_frag(k0 + 32, half), s0);
    v8f s1 = wmma_bf16(qf0, load_b_frag(k1, half), z);
    s1 = wmma_bf16(qf1, load_b_frag(k1 + 32, half), s1);

    // ---- causal mask (only tail steps; wave-uniform branch) ----
    if (kb + 32 > qb) {
#pragma unroll
      for (int r = 0; r < 8; ++r) {
        const int q = qb + r + 8 * half;
        if (kb + n16 > q)      s0[r] = -1e30f;
        if (kb + 16 + n16 > q) s1[r] = -1e30f;
      }
    }

    // ---- online softmax (row m = r + 8*half lives across 16 lanes) ----
    float p0[8], p1[8];
#pragma unroll
    for (int r = 0; r < 8; ++r) {
      float v = fmaxf(s0[r], s1[r]);
      v = fmaxf(v, __shfl_xor(v, 1));
      v = fmaxf(v, __shfl_xor(v, 2));
      v = fmaxf(v, __shfl_xor(v, 4));
      v = fmaxf(v, __shfl_xor(v, 8));
      const float mnew = fmaxf(mrow[r], v);
      const float corr = __expf(mrow[r] - mnew);
      const float a0 = __expf(s0[r] - mnew);
      const float a1 = __expf(s1[r] - mnew);
      float t = a0 + a1;
      t += __shfl_xor(t, 1);
      t += __shfl_xor(t, 2);
      t += __shfl_xor(t, 4);
      t += __shfl_xor(t, 8);
      lrow[r] = lrow[r] * corr + t;
      mrow[r] = mnew;
      p0[r] = a0; p1[r] = a1;
      acc[0][r] *= corr; acc[1][r] *= corr; acc[2][r] *= corr; acc[3][r] *= corr;
    }

    // ---- transpose P: C-layout f32 -> A-layout bf16 via per-wave LDS ----
    __bf16* pl = &lds[wave][0][0];
#pragma unroll
    for (int r = 0; r < 8; ++r) {
      pl[(r + 8 * half) * 32 + n16]      = (__bf16)p0[r];
      pl[(r + 8 * half) * 32 + 16 + n16] = (__bf16)p1[r];
    }
    asm volatile("s_wait_dscnt 0" ::: "memory");   // order cross-lane LDS use
    const v16bf pf = load_a_frag(pl + n16 * 32, half);

    // ---- ctx += P V (V transposed: 32B contiguous per lane) ----
#pragma unroll
    for (int dt = 0; dt < 4; ++dt) {
      const __bf16* vcol = Vtb + ((h << 6) + dt * 16 + n16) * SEQ + kb;
      acc[dt] = wmma_bf16(pf, load_b_frag(vcol, half), acc[dt]);
    }
  }

  // ---- normalize and write ctx bf16 [s][h*64 + d] ----
#pragma unroll
  for (int r = 0; r < 8; ++r) {
    const float inv = 1.0f / lrow[r];
    const int s = qb + r + 8 * half;
    __bf16* crow = ctxb + s * HID + (h << 6);
#pragma unroll
    for (int dt = 0; dt < 4; ++dt)
      crow[dt * 16 + n16] = (__bf16)(acc[dt][r] * inv);
  }
}

// ---------------------------------------------------------------------------
// Output projection: out = ctx @ out_w^T + out_b  (fp32 result).
// Grid: (4096/32)*(1024/64)/8 = 256 blocks of 256.
// ---------------------------------------------------------------------------
__global__ __launch_bounds__(256) void out_gemm(
    const __bf16* __restrict__ ctxb, const __bf16* __restrict__ owb,
    const float* __restrict__ out_b, float* __restrict__ out) {
  const int lane = threadIdx.x & 31, wave = threadIdx.x >> 5;
  const int n16 = lane & 15, half = lane >> 4;
  const int wid = blockIdx.x * 8 + wave;
  const int mt = wid / 16, nt = wid % 16;           // 128 x 16 tiles
  const int mbase = mt * 32, nbase = nt * 64;

  v8f acc[2][4] = {};
  for (int k = 0; k < HID; k += 32) {
    v16bf a0 = load_a_frag(ctxb + (mbase + n16) * HID + k, half);
    v16bf a1 = load_a_frag(ctxb + (mbase + 16 + n16) * HID + k, half);
#pragma unroll
    for (int j = 0; j < 4; ++j) {
      v16bf b = load_b_frag(owb + (nbase + j * 16 + n16) * HID + k, half);
      acc[0][j] = wmma_bf16(a0, b, acc[0][j]);
      acc[1][j] = wmma_bf16(a1, b, acc[1][j]);
    }
  }

#pragma unroll
  for (int j = 0; j < 4; ++j) {
    const int o = nbase + j * 16 + n16;
    const float bias = out_b[o];
#pragma unroll
    for (int mi = 0; mi < 2; ++mi) {
#pragma unroll
      for (int r = 0; r < 8; ++r) {
        const int s = mbase + mi * 16 + r + 8 * half;
        out[s * HID + o] = acc[mi][j][r] + bias;
      }
    }
  }
}

// ---------------------------------------------------------------------------
extern "C" void kernel_launch(void* const* d_in, const int* in_sizes, int n_in,
                              void* d_out, int out_size, void* d_ws, size_t ws_size,
                              hipStream_t stream) {
  (void)in_sizes; (void)n_in; (void)out_size; (void)ws_size;
  const float* x     = (const float*)d_in[0];   // [4096,1024]
  const float* qkv_w = (const float*)d_in[1];   // [3072,1024]
  const float* qkv_b = (const float*)d_in[2];   // [3072]
  const float* out_w = (const float*)d_in[3];   // [1024,1024]
  const float* out_b = (const float*)d_in[4];   // [1024]
  float* out = (float*)d_out;                   // [4096,1024] f32

  // Workspace carve (bf16 elements): 24M elems = 48 MB total.
  __bf16* ws   = (__bf16*)d_ws;
  __bf16* xb   = ws;                          // 4096*1024
  __bf16* wb   = xb  + (size_t)SEQ * HID;     // 3072*1024
  __bf16* owb  = wb  + (size_t)3 * HID * HID; // 1024*1024
  __bf16* Qb   = owb + (size_t)HID * HID;     // 16*4096*64
  __bf16* Kb   = Qb  + (size_t)NH * SEQ * HD;
  __bf16* Vtb  = Kb  + (size_t)NH * SEQ * HD;
  __bf16* ctxb = Vtb + (size_t)NH * SEQ * HD; // 4096*1024

  const int nx = SEQ * HID, nw = 3 * HID * HID, no = HID * HID;
  cvt_bf16<<<nx / 4 / 256, 256, 0, stream>>>(x, xb, nx);
  cvt_bf16<<<nw / 4 / 256, 256, 0, stream>>>(qkv_w, wb, nw);
  cvt_bf16<<<no / 4 / 256, 256, 0, stream>>>(out_w, owb, no);

  qkv_gemm<<<768, 256, 0, stream>>>(xb, wb, qkv_b, Qb, Kb, Vtb);
  flash_attn<<<512, 256, 0, stream>>>(Qb, Kb, Vtb, ctxb);
  out_gemm<<<256, 256, 0, stream>>>(ctxb, owb, out_b, out);
}
